// qGIN_52750788330061
// MI455X (gfx1250) — matrix operand order repaired and checked by reading, not compiled, not run on hardware
//
#include <hip/hip_runtime.h>

typedef _Float16 v16h __attribute__((ext_vector_type(16)));
typedef float    v8f  __attribute__((ext_vector_type(8)));

#define NN 50000
#define NE 800000
#define HD 64
#define NLAY 4
#define NG 512
#define NC 10

__global__ void k_zero(float* p, int n) {
    int i = blockIdx.x * blockDim.x + threadIdx.x;
    if (i < n) p[i] = 0.f;
}

// h[n][c] = b_emb[c] + sum_k x[n][k]*W[k][c] + pos[n][k]*W[3+k][c]; also zeroes agg
__global__ void k_embed(const float* __restrict__ x, const float* __restrict__ pos,
                        const float* __restrict__ W, const float* __restrict__ b,
                        float* __restrict__ h, float* __restrict__ agg) {
    int i = blockIdx.x * blockDim.x + threadIdx.x;
    if (i >= NN * HD) return;
    int n = i >> 6, c = i & 63;
    float acc = b[c];
#pragma unroll
    for (int k = 0; k < 3; ++k) acc += x[n * 3 + k] * W[k * HD + c];
#pragma unroll
    for (int k = 0; k < 3; ++k) acc += pos[n * 3 + k] * W[(3 + k) * HD + c];
    h[i] = acc;
    agg[i] = 0.f;
}

// agg[dst] += h[src]; 4 threads per edge, 16 floats each (4x b128 gather + 16 hw f32 atomics)
__global__ void k_scatter(const float* __restrict__ h, const int* __restrict__ ei,
                          float* __restrict__ agg) {
    int i = blockIdx.x * blockDim.x + threadIdx.x;
    if (i >= NE * 4) return;
    int e = i >> 2, q = (i & 3) * 16;
    int s = ei[e], d = ei[NE + e];
    const float4* hp = (const float4*)(h + (size_t)s * HD + q);
    float* o = agg + (size_t)d * HD + q;
#pragma unroll
    for (int c4 = 0; c4 < 4; ++c4) {
        float4 v = hp[c4];
        unsafeAtomicAdd(o + c4 * 4 + 0, v.x);
        unsafeAtomicAdd(o + c4 * 4 + 1, v.y);
        unsafeAtomicAdd(o + c4 * 4 + 2, v.z);
        unsafeAtomicAdd(o + c4 * 4 + 3, v.w);
    }
}

// Z[nrows,64] = A[nrows,64] @ W[64,64] via v_wmma_f32_16x16x32_f16, C init = inline 0.
// mode 1: A = (1+eps[layer])*A0 + A1          (GIN residual + aggregation)
// mode 2: A = relu(A0*scale[K] + shift[K])    (fused BN+ReLU on GEMM input; aux = scsh)
// mode 3: A = A0 / max(counts[row],1)         (head GEMM on pooled sums; aux = counts)
// No bias: modes 1/2 feed training-mode BatchNorm, which is exactly invariant to
// per-column constants; mode 3's bias is applied in k_head before the ReLU.
// Block = 128 threads = 4 waves; wave -> one 16-row M tile, 4 N tiles, K=64 (2 WMMAs each).
// W staged TRANSPOSED in LDS (Wt[n][k]): lane's B vector (contiguous k order: elem j <-> K=16*hi+j)
// is 32 contiguous 32B-aligned bytes -> 2x ds_load_b128.
// Z may alias A0/A1 (row-local). Block 0 also zeroes the stats buffer for the next colstats.
__global__ void k_gemm64(int mode, const float* A0, const float* A1,
                         const float* __restrict__ W, const float* __restrict__ aux,
                         const float* __restrict__ epsPtr, int layer,
                         float* Z, float* statsZero, int nrows) {
    __shared__ __align__(32) _Float16 Wt[64 * 64];  // Wt[n*64 + k]
    __shared__ __align__(32) float sc[64], sh[64];
    int tid = threadIdx.x;
    if (statsZero && blockIdx.x == 0 && tid < 128) statsZero[tid] = 0.f;
    for (int t = tid; t < 64 * 64; t += 128) {
        int k = t >> 6, n = t & 63;
        Wt[n * 64 + k] = (_Float16)W[t];            // coalesced global read
    }
    if (mode == 2 && tid < 64) { sc[tid] = aux[tid]; sh[tid] = aux[64 + tid]; }
    __syncthreads();

    int tile = blockIdx.x * 4 + (tid >> 5);         // wave-uniform
    if (tile * 16 >= nrows) return;                 // whole-wave skip: EXEC stays all-1s
    int lane = tid & 31, m = lane & 15, hi = lane >> 4;
    int row = tile * 16 + m;
    float epsf = (mode == 1) ? (1.0f + epsPtr[layer]) : 0.f;
    float rs   = (mode == 3) ? (1.0f / fmaxf(aux[row], 1.f)) : 1.f;

    // A operand: contiguous k order (elem s <-> K=16*hi+s); per lane 4 contiguous 8-float chunks,
    // each loaded as two explicit float4 (global_load_b128).
    v16h a[2];
#pragma unroll
    for (int kh = 0; kh < 2; ++kh) {
#pragma unroll
        for (int g = 0; g < 2; ++g) {
            int kb = kh * 32 + hi * 16 + g * 8;
            const float* p0 = A0 + (size_t)row * HD + kb;
            float4 x0 = *(const float4*)(p0);
            float4 x1 = *(const float4*)(p0 + 4);
            float v[8] = {x0.x, x0.y, x0.z, x0.w, x1.x, x1.y, x1.z, x1.w};
            if (mode == 1) {
                const float* p1 = A1 + (size_t)row * HD + kb;
                float4 y0 = *(const float4*)(p1);
                float4 y1 = *(const float4*)(p1 + 4);
                float w[8] = {y0.x, y0.y, y0.z, y0.w, y1.x, y1.y, y1.z, y1.w};
#pragma unroll
                for (int j = 0; j < 8; ++j) v[j] = epsf * v[j] + w[j];
            } else if (mode == 2) {
                float4 s0 = *(const float4*)&sc[kb];
                float4 s1 = *(const float4*)&sc[kb + 4];
                float4 t0 = *(const float4*)&sh[kb];
                float4 t1 = *(const float4*)&sh[kb + 4];
                float ss[8] = {s0.x, s0.y, s0.z, s0.w, s1.x, s1.y, s1.z, s1.w};
                float tt[8] = {t0.x, t0.y, t0.z, t0.w, t1.x, t1.y, t1.z, t1.w};
#pragma unroll
                for (int j = 0; j < 8; ++j) v[j] = fmaxf(v[j] * ss[j] + tt[j], 0.f);
            } else {
#pragma unroll
                for (int j = 0; j < 8; ++j) v[j] *= rs;
            }
#pragma unroll
            for (int j = 0; j < 8; ++j) a[kh][g * 8 + j] = (_Float16)v[j];
        }
    }

#pragma unroll
    for (int nt = 0; nt < 4; ++nt) {
        int col = nt * 16 + m;                      // C/D layout: col = lane&15
        v8f c = {};                                 // first WMMA uses inline-0 SRC2
#pragma unroll
        for (int kh = 0; kh < 2; ++kh) {
            // 16 contiguous halfs: Wt[col][kh*32 + hi*16 .. +15], 32B aligned
            v16h bm = *(const v16h*)&Wt[(size_t)col * 64 + kh * 32 + hi * 16];
            c = __builtin_amdgcn_wmma_f32_16x16x32_f16(false, a[kh], false, bm,
                                                       (short)0, c, false, false);
        }
#pragma unroll
        for (int r = 0; r < 8; ++r)                 // C/D: VGPR r <-> row 8*hi + r
            Z[(size_t)(tile * 16 + hi * 8 + r) * HD + col] = c[r];
    }
}

// column sums + sum-of-squares for BatchNorm (block LDS reduce -> 128 atomics/block)
__global__ void k_colstats(const float* __restrict__ Z, float* __restrict__ stats, int nrows) {
    __shared__ float s1[256], s2[256];
    int tid = threadIdx.x;
    int col = tid & 63;
    float a = 0.f, b = 0.f;
    for (int row = blockIdx.x * 4 + (tid >> 6); row < nrows; row += gridDim.x * 4) {
        float v = Z[(size_t)row * HD + col];
        a += v; b += v * v;
    }
    s1[tid] = a; s2[tid] = b;
    __syncthreads();
    if (tid < 64) {
        a = s1[tid] + s1[tid + 64] + s1[tid + 128] + s1[tid + 192];
        b = s2[tid] + s2[tid + 64] + s2[tid + 128] + s2[tid + 192];
        unsafeAtomicAdd(&stats[tid], a);
        unsafeAtomicAdd(&stats[64 + tid], b);
    }
}

__global__ void k_bnprep(const float* __restrict__ stats, const float* __restrict__ g,
                         const float* __restrict__ be, float* __restrict__ scsh, float invn) {
    int t = threadIdx.x;
    float mean = stats[t] * invn;
    float var  = stats[64 + t] * invn - mean * mean;
    float s = g[t] * rsqrtf(var + 1e-5f);
    scsh[t] = s;
    scsh[64 + t] = be[t] - mean * s;
}

// h = relu(BN(Z)); optionally zeroes agg for the next layer in the same pass
__global__ void k_bnrelu(const float* __restrict__ Z, const float* __restrict__ scsh,
                         float* __restrict__ h, float* aggZero) {
    int i = blockIdx.x * blockDim.x + threadIdx.x;
    if (i >= NN * HD) return;
    int c = i & 63;
    float v = Z[i];                                 // read before any aliased write
    h[i] = fmaxf(v * scsh[c] + scsh[64 + c], 0.f);
    if (aggZero) aggZero[i] = 0.f;
}

__global__ void k_pool(const float* __restrict__ h, const int* __restrict__ batch,
                       float* __restrict__ pooled, float* __restrict__ counts) {
    int i = blockIdx.x * blockDim.x + threadIdx.x;
    if (i >= NN * HD) return;
    int n = i >> 6, c = i & 63;
    int g = batch[n];
    unsafeAtomicAdd(&pooled[g * HD + c], h[i]);
    if (c == 0) unsafeAtomicAdd(&counts[g], 1.f);
}

// out[g][cls] = b2[cls] + sum_k relu(Zh[g][k] + b1[k]) * W2[k][cls]; also bit_sum = 0
__global__ void k_head(const float* __restrict__ Zh, const float* __restrict__ b1,
                       const float* __restrict__ W2, const float* __restrict__ b2,
                       float* __restrict__ out) {
    int i = blockIdx.x * blockDim.x + threadIdx.x;
    if (i == 0) out[NG * NC] = 0.f;
    if (i >= NG * NC) return;
    int g = i / NC, cl = i % NC;
    float acc = b2[cl];
#pragma unroll 4
    for (int k = 0; k < HD; ++k)
        acc += fmaxf(Zh[g * HD + k] + b1[k], 0.f) * W2[k * NC + cl];
    out[i] = acc;
}

extern "C" void kernel_launch(void* const* d_in, const int* in_sizes, int n_in,
                              void* d_out, int out_size, void* d_ws, size_t ws_size,
                              hipStream_t stream) {
    const float* x     = (const float*)d_in[0];
    const float* pos   = (const float*)d_in[1];
    const int*   ei    = (const int*)d_in[2];
    const int*   batch = (const int*)d_in[3];
    const float* Wemb  = (const float*)d_in[4];
    const float* bemb  = (const float*)d_in[5];
    const float* eps   = (const float*)d_in[6];
    const float* W1    = (const float*)d_in[7];
    const float* g1    = (const float*)d_in[9];
    const float* be1   = (const float*)d_in[10];
    const float* W2    = (const float*)d_in[11];
    const float* g2    = (const float*)d_in[13];
    const float* be2   = (const float*)d_in[14];
    const float* Wl1   = (const float*)d_in[15];
    const float* bl1   = (const float*)d_in[16];
    const float* Wl2   = (const float*)d_in[17];
    const float* bl2   = (const float*)d_in[18];
    float* out = (float*)d_out;

    char* ws = (char*)d_ws;                       // all offsets 512B-aligned
    float* agg    = (float*)(ws + 0);             // 12.8 MB, also reused as Z
    float* h      = (float*)(ws + 12800000);      // 12.8 MB
    float* stats  = (float*)(ws + 25600000);      // 128 f
    float* scsh   = (float*)(ws + 25600512);      // 128 f
    float* pooled = (float*)(ws + 25601024);      // 512*64 f
    float* counts = (float*)(ws + 25732096);      // 512 f (contiguous after pooled)
    float* Zh     = (float*)(ws + 25734144);      // 512*64 f

    const int TPB = 256;
    const int gridNNHD = (NN * HD + TPB - 1) / TPB;        // 12500
    const int tilesNN  = NN / 16;                           // 3125
    const int gemmGrid = (tilesNN + 3) / 4;                 // 782 blocks of 4 waves

    k_embed<<<gridNNHD, TPB, 0, stream>>>(x, pos, Wemb, bemb, h, agg);

    for (int l = 0; l < NLAY; ++l) {
        k_scatter<<<(NE * 4 + TPB - 1) / TPB, TPB, 0, stream>>>(h, ei, agg);

        // z1 = ((1+eps)h + agg) @ W1   (Z aliases agg: row-local, safe; zeroes stats)
        k_gemm64<<<gemmGrid, 128, 0, stream>>>(1, h, agg, W1 + l * HD * HD,
                                               nullptr, eps, l, agg, stats, NN);
        k_colstats<<<256, 256, 0, stream>>>(agg, stats, NN);
        k_bnprep<<<1, 64, 0, stream>>>(stats, g1 + l * HD, be1 + l * HD, scsh, 1.f / NN);

        // z2 = relu(BN(z1)) @ W2  (BN+ReLU fused into A read; zeroes stats again)
        k_gemm64<<<gemmGrid, 128, 0, stream>>>(2, agg, nullptr, W2 + l * HD * HD,
                                               scsh, nullptr, 0, agg, stats, NN);
        k_colstats<<<256, 256, 0, stream>>>(agg, stats, NN);
        k_bnprep<<<1, 64, 0, stream>>>(stats, g2 + l * HD, be2 + l * HD, scsh, 1.f / NN);

        // h = relu(BN(z2)); zero agg for next layer's scatter in the same pass
        k_bnrelu<<<gridNNHD, TPB, 0, stream>>>(agg, scsh, h,
                                               (l + 1 < NLAY) ? agg : nullptr);
    }

    // global mean pool per graph (sums + counts; division fused into head GEMM)
    k_zero<<<(NG * HD + NG + TPB - 1) / TPB, TPB, 0, stream>>>(pooled, NG * HD + NG);
    k_pool<<<gridNNHD, TPB, 0, stream>>>(h, batch, pooled, counts);

    // head: (pooled/counts) @ W_lin1 (WMMA, 32 tiles), then bias+relu+64->10 proj
    k_gemm64<<<(NG / 16 + 3) / 4, 128, 0, stream>>>(3, pooled, nullptr, Wl1,
                                                    counts, nullptr, 0, Zh, nullptr, NG);
    k_head<<<(NG * NC + TPB - 1) / TPB, TPB, 0, stream>>>(Zh, bl1, Wl2, bl2, out);
}
